// RFFTCirculant1D_86887188398202
// MI455X (gfx1250) — matrix operand order repaired and checked by reading, not compiled, not run on hardware
//
#include <hip/hip_runtime.h>
#include <hip/hip_bf16.h>

// ---------------------------------------------------------------------------
// y = irfft(rfft(x) * H_masked) + bias  ==  circulant GEMM:
//   y[b,n] = sum_m x[b,m] * c[(n-m) & 4095] + bias[n]
// bf16 WMMA GEMM on gfx1250 (wave32): double-buffered LDS pipeline staged by
// TDM tensor_load_to_lds (TENSORcnt) / GLOBAL_LOAD_ASYNC_TO_LDS (ASYNCcnt).
// If workspace allows, x is pre-converted to bf16 so the GEMM hot loop is
// pure DMA + WMMA (no conversion VALU).
// ---------------------------------------------------------------------------

#define N_FEAT 4096
#define BATCH  8192
#define K_KEEP 1024

#define BM 256
#define BN 128
#define BK 32
#define NP (N_FEAT / BK)   // 128 K-panels
#define LDA 40             // (BK + 8) u16 stride: conflict-free fragment reads
#define LDB 40

#define W_BYTES   ((size_t)N_FEAT * N_FEAT * 2)          // 33.5 MB
#define XBF_BYTES ((size_t)BATCH  * N_FEAT * 2)          // 67 MB
#define WS_SMALL  (16384 + W_BYTES)
#define WS_BIG    (16384 + W_BYTES + XBF_BYTES)

typedef __attribute__((ext_vector_type(8)))  float        v8f;
typedef __attribute__((ext_vector_type(8)))  unsigned int v8u;
typedef __attribute__((ext_vector_type(16))) __bf16       v16bf;
typedef __attribute__((ext_vector_type(2)))  __bf16       v2bf;

#if defined(__has_builtin)
#if __has_builtin(__builtin_amdgcn_global_load_async_to_lds_b128) && \
    __has_builtin(__builtin_amdgcn_s_wait_asynccnt)
#define HAVE_ASYNC_LDS 1
#endif
#if __has_builtin(__builtin_amdgcn_tensor_load_to_lds) && \
    __has_builtin(__builtin_amdgcn_s_wait_tensorcnt)
#define HAVE_TDM 1
#endif
#endif

#ifdef HAVE_ASYNC_LDS
// Builtin signature (from hipcc diagnostic): pointee is GCC generic-vector int4.
typedef int b128_t __attribute__((vector_size(16)));
#define AS1P(p) ((__attribute__((address_space(1))) b128_t*)(uintptr_t)(p))
#define AS3P(p) ((__attribute__((address_space(3))) b128_t*)(unsigned)(uintptr_t)(p))
#endif

// two fp32 -> packed bf16; native bf16 converts (v_cvt_pk_bf16_f32) on gfx1250
__device__ __forceinline__ unsigned int pack_bf16(float lo, float hi) {
    v2bf v;
    v.x = (__bf16)lo;
    v.y = (__bf16)hi;
    return __builtin_bit_cast(unsigned int, v);
}

#ifdef HAVE_TDM
typedef unsigned int u32x4 __attribute__((vector_size(16)));
typedef int          i32x4 __attribute__((vector_size(16)));
typedef int          i32x8 __attribute__((vector_size(32)));

// 2D TDM tile: `rows` rows x 64B (32 x bf16) with row stride 4096 elems (2B),
// landing in LDS with 16B pad after every 64B -> LDS row stride 80B (40 u16).
__device__ __forceinline__ void tdm_load_tile(const unsigned short* gsrc,
                                              unsigned short* ldst,
                                              int rows, int tensor_rows) {
    unsigned long long ga = (unsigned long long)(uintptr_t)gsrc;
    unsigned int       la = (unsigned int)(uintptr_t)ldst;  // LDS byte address
    u32x4 g0 = {0u, 0u, 0u, 0u};
    g0[0] = 1u;                                         // count=1 (valid user D#)
    g0[1] = la;                                         // lds_addr   [63:32]
    g0[2] = (unsigned int)(ga & 0xFFFFFFFFu);           // global_addr[95:64]
    g0[3] = (unsigned int)((ga >> 32) & 0x01FFFFFFu)    // global_addr[120:96]
          | (2u << 30);                                 // type=2 ("image")
    i32x8 g1 = {0, 0, 0, 0, 0, 0, 0, 0};
    g1[0] = (1 << 16)                                   // data_size = 2 bytes
          | (1 << 20)                                   // pad_enable
          | (3 << 22)                                   // pad_interval: 16 DWORDs (64B)
          | (3 << 25);                                  // pad_amount:   4 DWORDs (16B)
    g1[1] = (int)((unsigned)N_FEAT << 16);              // tensor_dim0[15:0]
    g1[2] = (int)(((unsigned)tensor_rows & 0xFFFFu) << 16); // dim0 hi=0 | dim1 lo16
    g1[3] = (int)(((unsigned)tensor_rows >> 16) & 0xFFFFu)  // dim1 hi16
          | (int)(32u << 16);                           // tile_dim0 = 32 elems
    g1[4] = rows;                                       // tile_dim1, tile_dim2=0
    g1[5] = N_FEAT;                                     // tensor_dim0_stride[31:0]
    i32x4 z4 = {0, 0, 0, 0};
#if __clang_major__ >= 23
    i32x8 z8 = {0, 0, 0, 0, 0, 0, 0, 0};
    __builtin_amdgcn_tensor_load_to_lds(g0, g1, z4, z4, z8, 0);
#else
    __builtin_amdgcn_tensor_load_to_lds(g0, g1, z4, z4, 0);
#endif
}
#endif

// ---------------------------------------------------------------------------
// Kernel 1: circulant kernel c[0..4095] (fp32)
// ---------------------------------------------------------------------------
__global__ void build_c_kernel(const float* __restrict__ Hr,
                               const float* __restrict__ Hi,
                               float* __restrict__ c) {
    int t = blockIdx.x * blockDim.x + threadIdx.x;     // 0..4095
    float acc = Hr[0];
    const float w = 6.28318530717958647692f / (float)N_FEAT;
    for (int k = 1; k < K_KEEP; ++k) {
        int ph = (k * t) & (N_FEAT - 1);
        float s, co;
        __sincosf(w * (float)ph, &s, &co);
        acc += 2.0f * (Hr[k] * co - Hi[k] * s);
    }
    c[t] = acc * (1.0f / (float)N_FEAT);
}

// ---------------------------------------------------------------------------
// Kernel 2: Wt_bf16[n][m] = c[(n - m) & 4095]  (transposed layout: row n is
// contiguous in m, so GEMM B-panels are plain 2D tiles -> TDM/async DMA)
// ---------------------------------------------------------------------------
__global__ void build_w_kernel(const float* __restrict__ c,
                               unsigned short* __restrict__ Wt) {
    int base = (blockIdx.x * blockDim.x + threadIdx.x) * 8;   // 8 elems / thread
    int n = base >> 12;
    int m = base & (N_FEAT - 1);
    float v[8];
#pragma unroll
    for (int i = 0; i < 8; ++i)
        v[i] = c[(n - m - i) & (N_FEAT - 1)];
    uint4 pk;
    pk.x = pack_bf16(v[0], v[1]);
    pk.y = pack_bf16(v[2], v[3]);
    pk.z = pack_bf16(v[4], v[5]);
    pk.w = pack_bf16(v[6], v[7]);
    *(uint4*)(Wt + base) = pk;
}

// ---------------------------------------------------------------------------
// Kernel 2b: pre-convert x (fp32) -> xbf (bf16), 8 elems/thread
// ---------------------------------------------------------------------------
__global__ void build_xbf_kernel(const float* __restrict__ x,
                                 unsigned short* __restrict__ xbf) {
    size_t base = ((size_t)blockIdx.x * blockDim.x + threadIdx.x) * 8;
    const float4* p = (const float4*)(x + base);
    float4 f0 = p[0], f1 = p[1];
    uint4 pk;
    pk.x = pack_bf16(f0.x, f0.y);
    pk.y = pack_bf16(f0.z, f0.w);
    pk.z = pack_bf16(f1.x, f1.y);
    pk.w = pack_bf16(f1.z, f1.w);
    *(uint4*)(xbf + base) = pk;
}

// ---------------------------------------------------------------------------
// Kernel 3: y[8192,4096] = x @ W + bias  (bf16 WMMA, double-buffered)
// 256 threads = 8 waves; block tile 256x128; wave tile 64x64; BK=32.
// PRECONV=true : both panels staged by TDM/async DMA (no conversion VALU).
// PRECONV=false: A converted fp32->bf16 in-kernel while WMMAs run.
// ---------------------------------------------------------------------------
template <bool PRECONV>
__global__ void __launch_bounds__(256)
circulant_gemm_kernel(const float* __restrict__ x,
                      const unsigned short* __restrict__ xbf,
                      const unsigned short* __restrict__ Wt,
                      const float* __restrict__ bias,
                      float* __restrict__ y) {
    __shared__ unsigned short sA[2][BM * LDA];   // [m][k] bf16 bits
    __shared__ unsigned short sB[2][BN * LDB];   // [n][k] bf16 bits
    __shared__ float sBias[BN];

    const int t     = threadIdx.x;
    const int lane  = t & 31;
    const int wv    = t >> 5;          // 0..7
    const int wr    = wv & 3;          // wave row -> 64 rows each
    const int wc    = wv >> 2;         // wave col -> 64 cols each
    const int lrow  = lane & 15;
    const int lhalf = lane >> 4;

    const int n0 = blockIdx.x * BN;
    const int m0 = blockIdx.y * BM;

    const int rowB = t >> 1;
    const int kB   = (t & 1) * 16;

    const float*          xrow   = x   + (size_t)(m0 + t) * N_FEAT;
    const unsigned short* xbfrow = xbf + (size_t)(m0 + t) * N_FEAT;
    const unsigned short* wtrow  = Wt  + (size_t)(n0 + rowB) * N_FEAT + kB;
    unsigned int* sAdst0 = (unsigned int*)&sA[0][t * LDA];
    unsigned int* sAdst1 = (unsigned int*)&sA[1][t * LDA];

    const v8f vzero = {0.f, 0.f, 0.f, 0.f, 0.f, 0.f, 0.f, 0.f};
    v8f acc[4][4];
#pragma unroll
    for (int r = 0; r < 4; ++r)
#pragma unroll
        for (int j = 0; j < 4; ++j) acc[r][j] = vzero;

    // ---- panel staging: TDM descriptor DMA (wave 0) > per-lane async > sync
    auto copyB = [&](unsigned short* dstBase, int k0) {
#if defined(HAVE_TDM)
        if (t < 32) tdm_load_tile(Wt + (size_t)n0 * N_FEAT + k0, dstBase, BN, N_FEAT);
#elif defined(HAVE_ASYNC_LDS)
        const unsigned short* src = wtrow + k0;
        unsigned short*       dst = dstBase + rowB * LDB + kB;
        __builtin_amdgcn_global_load_async_to_lds_b128(AS1P(src),     AS3P(dst),     0, 0);
        __builtin_amdgcn_global_load_async_to_lds_b128(AS1P(src + 8), AS3P(dst + 8), 0, 0);
#else
        const unsigned short* src = wtrow + k0;
        unsigned short*       dst = dstBase + rowB * LDB + kB;
        *(uint4*)dst       = *(const uint4*)src;
        *((uint4*)dst + 1) = *((const uint4*)src + 1);
#endif
    };
    auto copyA_bf = [&](unsigned short* dstBase, int k0) {   // PRECONV only
#if defined(HAVE_TDM)
        if (t < 32) tdm_load_tile(xbf + (size_t)m0 * N_FEAT + k0, dstBase, BM, BATCH);
#elif defined(HAVE_ASYNC_LDS)
        const unsigned short* src = xbfrow + k0;
        unsigned short*       dst = dstBase + t * LDA;
#pragma unroll
        for (int q = 0; q < 4; ++q)
            __builtin_amdgcn_global_load_async_to_lds_b128(AS1P(src + 8 * q),
                                                           AS3P(dst + 8 * q), 0, 0);
#else
        const unsigned short* src = xbfrow + k0;
        unsigned short*       dst = dstBase + t * LDA;
#pragma unroll
        for (int q = 0; q < 4; ++q)
            *((uint4*)dst + q) = *((const uint4*)src + q);
#endif
    };
    auto waitStage = [&]() {
#if defined(HAVE_TDM)
        if (t < 32) __builtin_amdgcn_s_wait_tensorcnt(0);
#elif defined(HAVE_ASYNC_LDS)
        __builtin_amdgcn_s_wait_asynccnt(0);
#endif
    };
    auto loadA = [&](float4* fa, int k0) {
        const float4* p = (const float4*)(xrow + k0);
#pragma unroll
        for (int q = 0; q < 8; ++q) fa[q] = p[q];
    };
    auto storeA = [&](unsigned int* dst, const float4* fa) {
#pragma unroll
        for (int q = 0; q < 8; ++q) {
            dst[2 * q + 0] = pack_bf16(fa[q].x, fa[q].y);
            dst[2 * q + 1] = pack_bf16(fa[q].z, fa[q].w);
        }
    };

    // ---- prolog: stage panel 0 into buffer 0, stage bias to LDS ----
    float4 fa[8];
    copyB(sB[0], 0);
    if constexpr (PRECONV) {
        copyA_bf(sA[0], 0);
    } else {
        loadA(fa, 0);
        storeA(sAdst0, fa);
    }
#ifdef HAVE_ASYNC_LDS
    if (t < 32)
        __builtin_amdgcn_global_load_async_to_lds_b128(
            AS1P(bias + n0 + t * 4), AS3P(&sBias[t * 4]), 0, 0);
    __builtin_amdgcn_s_wait_asynccnt(0);
#else
    if (t < BN) sBias[t] = bias[n0 + t];
#endif
    waitStage();
    __syncthreads();

    for (int kp = 0; kp < NP; ++kp) {
        const int  cur  = kp & 1;
        const bool more = (kp + 1) < NP;
        const int  kn   = (kp + 1) * BK;

        // ---- issue next panel while this one computes ----
        if (more) {
            copyB(sB[cur ^ 1], kn);
            if constexpr (PRECONV) {
                copyA_bf(sA[cur ^ 1], kn);
            } else {
                loadA(fa, kn);
                if (kp + 2 < NP)
                    __builtin_prefetch(xrow + kn + BK, 0, 0);   // global_prefetch_b8
            }
        }

        // ---- fragments + 16 WMMAs on current buffer ----
        const unsigned short* bufA = sA[cur];
        const unsigned short* bufB = sB[cur];

        v16bf afrag[4];
#pragma unroll
        for (int r = 0; r < 4; ++r) {
            const int rb = (wr * 64 + r * 16 + lrow) * LDA;
            v8u au;
#pragma unroll
            for (int v = 0; v < 8; ++v) {
                // ISA 16-bit A 16x32 layout: VGPR v<4 -> K = 2v + 8*half,
                //                            v>=4    -> K = 16 + 2(v-4) + 8*half
                const int kpair = ((v < 4) ? (2 * v) : (8 + 2 * v)) + 8 * lhalf;
                au[v] = *(const unsigned int*)&bufA[rb + kpair];
            }
            afrag[r] = __builtin_bit_cast(v16bf, au);
        }

#pragma unroll
        for (int j = 0; j < 4; ++j) {
            const int cb = (wc * 64 + j * 16 + lrow) * LDB + lhalf * 8;
            v8u bu;
#pragma unroll
            for (int v = 0; v < 8; ++v)
                bu[v] = *(const unsigned int*)&bufB[cb + ((v < 4) ? (2 * v) : (8 + 2 * v))];
            v16bf bfrag = __builtin_bit_cast(v16bf, bu);
#pragma unroll
            for (int r = 0; r < 4; ++r) {
                acc[r][j] = __builtin_amdgcn_wmma_f32_16x16x32_bf16(
                    false, afrag[r], false, bfrag, (short)0, acc[r][j],
                    false, false);
            }
        }

        // ---- land next panel into the other buffer, then rendezvous ----
        if (more) {
            if constexpr (!PRECONV)
                storeA(cur ? sAdst0 : sAdst1, fa);
            waitStage();
            __syncthreads();
        }
    }

    // ---- epilogue: f32 C/D layout -> global + bias (bias from LDS) ----
#pragma unroll
    for (int r = 0; r < 4; ++r) {
#pragma unroll
        for (int j = 0; j < 4; ++j) {
            const int colOff = wc * 64 + j * 16 + lrow;
            const float b = sBias[colOff];
#pragma unroll
            for (int i = 0; i < 8; ++i) {
                const int row = m0 + wr * 64 + r * 16 + lhalf * 8 + i;
                y[(size_t)row * N_FEAT + n0 + colOff] = acc[r][j][i] + b;
            }
        }
    }
}

// ---------------------------------------------------------------------------
extern "C" void kernel_launch(void* const* d_in, const int* in_sizes, int n_in,
                              void* d_out, int out_size, void* d_ws, size_t ws_size,
                              hipStream_t stream) {
    const float* x  = (const float*)d_in[0];
    const float* Hr = (const float*)d_in[1];
    const float* Hi = (const float*)d_in[2];
    const float* bs = (const float*)d_in[3];
    float* y = (float*)d_out;

    // workspace layout: [0,16KB) c_f32 ; then Wt_bf16 (33.5MB) ; then xbf (67MB)
    float*          c   = (float*)d_ws;
    unsigned short* Wt  = (unsigned short*)((char*)d_ws + 16384);
    unsigned short* xbf = (unsigned short*)((char*)d_ws + 16384 + W_BYTES);

    build_c_kernel<<<N_FEAT / 256, 256, 0, stream>>>(Hr, Hi, c);
    build_w_kernel<<<(N_FEAT * N_FEAT) / (256 * 8), 256, 0, stream>>>(c, Wt);

    dim3 grid(N_FEAT / BN, BATCH / BM);
    if (ws_size >= WS_BIG) {
        build_xbf_kernel<<<(int)(((size_t)BATCH * N_FEAT) / (256 * 8)), 256, 0, stream>>>(x, xbf);
        circulant_gemm_kernel<true><<<grid, 256, 0, stream>>>(x, xbf, Wt, bs, y);
    } else {
        circulant_gemm_kernel<false><<<grid, 256, 0, stream>>>(x, xbf, Wt, bs, y);
    }

    (void)in_sizes; (void)n_in; (void)out_size;
}
